// HierarchicalFusionModel_50036368999056
// MI455X (gfx1250) — matrix operand order repaired and checked by reading, not compiled, not run on hardware
//
#include <hip/hip_runtime.h>
#include <hip/hip_bf16.h>

// ---------------------------------------------------------------------------
// HierarchicalFusionModel on MI455X (gfx1250), wave32 + WMMA f16->f32.
// B=128, T=2048, STATE=12, FAST=4, SLOW=4, FUSED=32, HIDDEN=128.
// All dense layers run as 16-token tiles through v_wmma_f32_16x16x32_f16,
// activations held in LDS per wave; weights pre-converted once into f16
// fragments in the exact CDNA5 B-matrix lane/half layout (ISA 7.12.2).
// ---------------------------------------------------------------------------

typedef __attribute__((ext_vector_type(16))) _Float16 v16h;
typedef __attribute__((ext_vector_type(8)))  float    v8f;

#define B_    128
#define T_    2048
#define NTOK_ (B_ * T_)   // 262144 tokens

union Frag16 { v16h v; _Float16 h[16]; uint4 q[2]; };

__device__ __forceinline__ v8f wmma_f16(v16h a, v16h b, v8f c) {
  // (neg_a, A, neg_b, B, c_mod, C, reuse_a, reuse_b)
  return __builtin_amdgcn_wmma_f32_16x16x32_f16(false, a, false, b, (short)0, c,
                                                false, false);
}

// A-matrix 16x32 f16 per ISA: lane L -> M=L%16, g=L/16; halves 0..7 hold
// K=8g..8g+7, halves 8..15 hold K=16+8g..16+8g+7 (two contiguous 16B reads).
__device__ __forceinline__ v16h load_A_frag_lds(const _Float16* base, int stride,
                                                int lane, int k0) {
  int m = lane & 15, g = lane >> 4;
  const _Float16* row = base + m * stride + k0 + 8 * g;
  Frag16 f;
  f.q[0] = *reinterpret_cast<const uint4*>(row);
  f.q[1] = *reinterpret_cast<const uint4*>(row + 16);
  return f.v;
}

// B fragments are stored pre-swizzled: element (lane, h) at frag[lane*16+h].
__device__ __forceinline__ v16h load_B_frag(const _Float16* frag, int lane) {
  const uint4* p = reinterpret_cast<const uint4*>(frag);
  Frag16 f;
  f.q[0] = p[lane * 2 + 0];
  f.q[1] = p[lane * 2 + 1];
  return f.v;
}

// ---------------------------------------------------------------------------
// Weight prep: W (dout x din, row-major f32, used as x @ W^T) -> f16 B-frags.
// B-matrix 32x16 f16 layout: lane L -> N=L%16, g=L/16; half h<8 -> K=8g+h, h>=8 -> K=16+8g+(h-8).
// Fragment id = nt*Kt + kt, each fragment 512 halves. grid=(Nt,Kt), block=32.
// ---------------------------------------------------------------------------
__global__ void k_prep(const float* __restrict__ W, int dout, int din,
                       _Float16* __restrict__ dst, int Kt) {
  int nt = blockIdx.x, kt = blockIdx.y, lane = threadIdx.x;
  int n = nt * 16 + (lane & 15);
  int kbase = kt * 32 + 8 * (lane >> 4);
  _Float16* out = dst + (size_t)(nt * Kt + kt) * 512 + lane * 16;
#pragma unroll
  for (int h = 0; h < 16; ++h) {
    int k = kbase + (h < 8 ? h : h + 8);
    float v = (n < dout && k < din) ? W[(size_t)n * din + k] : 0.0f;
    out[h] = (_Float16)v;
  }
}

__global__ void k_zero(float* p) { if (threadIdx.x == 0) *p = 0.0f; }

// ---------------------------------------------------------------------------
// Fast encoder + reparameterization + KL. One wave = one 16-token tile.
// occ[tok][12] -> relu(L1:12->128) -> L2:128->8 -> mu/lv -> z, kl.
// ---------------------------------------------------------------------------
__global__ void __launch_bounds__(32)
k_fast_enc(const float* __restrict__ occ, const float* __restrict__ eps,
           const _Float16* __restrict__ wf1, const float* __restrict__ b1,
           const _Float16* __restrict__ wf2, const float* __restrict__ b2,
           float* __restrict__ mu_out, float* __restrict__ z_out,
           float* __restrict__ kl_acc) {
  __shared__ __align__(16) _Float16 sIn[16 * 32];
  __shared__ __align__(16) _Float16 sAct[16 * 136];
  __shared__ float sOut[16 * 8];
  int lane = threadIdx.x;
  long tok0 = (long)blockIdx.x * 16;

  for (int idx = lane; idx < 16 * 32; idx += 32) {
    int m = idx >> 5, f = idx & 31;
    float v = (f < 12) ? occ[(tok0 + m) * 12 + f] : 0.0f;
    sIn[m * 32 + f] = (_Float16)v;
  }
  __syncthreads();

  v16h aIn = load_A_frag_lds(sIn, 32, lane, 0);
#pragma unroll
  for (int nt = 0; nt < 8; ++nt) {
    v8f c = {};
    c = wmma_f16(aIn, load_B_frag(wf1 + nt * 512, lane), c);
    int n = nt * 16 + (lane & 15), g = lane >> 4;
    float bb = b1[n];
#pragma unroll
    for (int r = 0; r < 8; ++r) {
      float v = c[r] + bb;
      sAct[(r + 8 * g) * 136 + n] = (_Float16)(v > 0.0f ? v : 0.0f);
    }
  }
  __syncthreads();

  {
    v8f c = {};
#pragma unroll
    for (int kt = 0; kt < 4; ++kt)
      c = wmma_f16(load_A_frag_lds(sAct, 136, lane, kt * 32),
                   load_B_frag(wf2 + kt * 512, lane), c);
    int n = lane & 15, g = lane >> 4;
    if (n < 8) {
      float bb = b2[n];
#pragma unroll
      for (int r = 0; r < 8; ++r) sOut[(r + 8 * g) * 8 + n] = c[r] + bb;
    }
  }
  __syncthreads();

  float klp = 0.0f;
#pragma unroll
  for (int idx = lane; idx < 64; idx += 32) {
    int m = idx >> 2, j = idx & 3;
    long tok = tok0 + m;
    float mu = sOut[m * 8 + j];
    float lv = sOut[m * 8 + 4 + j];
    float e  = eps[tok * 4 + j];
    float el = expf(lv);
    mu_out[tok * 4 + j] = mu;
    z_out[tok * 4 + j]  = mu + expf(0.5f * lv) * e;
    klp += -0.5f * (1.0f + lv - mu * mu - el);
  }
  for (int off = 16; off > 0; off >>= 1) klp += __shfl_down(klp, off, 32);
  if (lane == 0) atomicAdd(kl_acc, klp * (1.0f / (float)NTOK_));
}

// ---------------------------------------------------------------------------
// Slow-channel encoder: tokens (b, s) over strided occ[b, s*K], 12->128->4.
// ---------------------------------------------------------------------------
__global__ void __launch_bounds__(32)
k_slow_enc(const float* __restrict__ occ,
           const _Float16* __restrict__ wf1, const float* __restrict__ b1,
           const _Float16* __restrict__ wf2, const float* __restrict__ b2,
           float* __restrict__ sin_out, const int* __restrict__ kptr) {
  __shared__ __align__(16) _Float16 sIn[16 * 32];
  __shared__ __align__(16) _Float16 sAct[16 * 136];
  int K = *kptr;
  int S = (T_ + K - 1) / K;
  long nTokS = (long)B_ * S;
  int lane = threadIdx.x;
  long tok0 = (long)blockIdx.x * 16;
  if (tok0 >= nTokS) return;

  for (int idx = lane; idx < 16 * 32; idx += 32) {
    int m = idx >> 5, f = idx & 31;
    long tok = tok0 + m;
    float v = 0.0f;
    if (f < 12 && tok < nTokS) {
      long b = tok / S, s = tok % S;
      v = occ[(b * T_ + s * (long)K) * 12 + f];
    }
    sIn[m * 32 + f] = (_Float16)v;
  }
  __syncthreads();

  v16h aIn = load_A_frag_lds(sIn, 32, lane, 0);
#pragma unroll
  for (int nt = 0; nt < 8; ++nt) {
    v8f c = {};
    c = wmma_f16(aIn, load_B_frag(wf1 + nt * 512, lane), c);
    int n = nt * 16 + (lane & 15), g = lane >> 4;
    float bb = b1[n];
#pragma unroll
    for (int r = 0; r < 8; ++r) {
      float v = c[r] + bb;
      sAct[(r + 8 * g) * 136 + n] = (_Float16)(v > 0.0f ? v : 0.0f);
    }
  }
  __syncthreads();

  v8f c = {};
#pragma unroll
  for (int kt = 0; kt < 4; ++kt)
    c = wmma_f16(load_A_frag_lds(sAct, 136, lane, kt * 32),
                 load_B_frag(wf2 + kt * 512, lane), c);
  int n = lane & 15, g = lane >> 4;
  if (n < 4) {
    float bb = b2[n];
#pragma unroll
    for (int r = 0; r < 8; ++r) {
      long tok = tok0 + r + 8 * g;
      if (tok < nTokS) sin_out[tok * 4 + n] = c[r] + bb;
    }
  }
}

// ---------------------------------------------------------------------------
// GRU scan over S slow steps, hidden=4. One thread per batch element.
// ---------------------------------------------------------------------------
__global__ void __launch_bounds__(128)
k_gru(const float* __restrict__ sin_in, const float* __restrict__ Wih,
      const float* __restrict__ Whh, const float* __restrict__ bih,
      const float* __restrict__ bhh, float* __restrict__ hs,
      const int* __restrict__ kptr) {
  __shared__ float w[120];  // Wih(48) | Whh(48) | bih(12) | bhh(12)
  int tid = threadIdx.x;
  if (tid < 48) { w[tid] = Wih[tid]; w[48 + tid] = Whh[tid]; }
  if (tid < 12) { w[96 + tid] = bih[tid]; w[108 + tid] = bhh[tid]; }
  __syncthreads();
  int K = *kptr;
  int S = (T_ + K - 1) / K;
  long b = tid;
  float h[4] = {0.f, 0.f, 0.f, 0.f};
  for (int s = 0; s < S; ++s) {
    const float* x = &sin_in[(b * S + s) * 4];
    float xl[4] = {x[0], x[1], x[2], x[3]};
    float gi[12], gh[12];
#pragma unroll
    for (int i = 0; i < 12; ++i) {
      float a = w[96 + i], c = w[108 + i];
#pragma unroll
      for (int j = 0; j < 4; ++j) {
        a += w[i * 4 + j] * xl[j];
        c += w[48 + i * 4 + j] * h[j];
      }
      gi[i] = a; gh[i] = c;
    }
#pragma unroll
    for (int j = 0; j < 4; ++j) {
      float r = 1.0f / (1.0f + expf(-(gi[j] + gh[j])));
      float z = 1.0f / (1.0f + expf(-(gi[4 + j] + gh[4 + j])));
      float n = tanhf(gi[8 + j] + r * gh[8 + j]);
      h[j] = (1.0f - z) * n + z * h[j];
    }
#pragma unroll
    for (int j = 0; j < 4; ++j) hs[(b * S + s) * 4 + j] = h[j];
  }
}

// Expand slow states: h_seq[b][t][j] = hs[b][t/K][j]
__global__ void k_expand(const float* __restrict__ hs, float* __restrict__ out,
                         const int* __restrict__ kptr) {
  int K = *kptr;
  int S = (T_ + K - 1) / K;
  long idx = (long)blockIdx.x * blockDim.x + threadIdx.x;
  if (idx >= (long)NTOK_ * 4) return;
  int j = idx & 3;
  long bt = idx >> 2;
  int t = (int)(bt % T_);
  long b = bt / T_;
  out[idx] = hs[(b * S + t / K) * 4 + j];
}

// ---------------------------------------------------------------------------
// Fusion -> predictor -> decoder, fully fused per 16-token tile.
// ---------------------------------------------------------------------------
__device__ __forceinline__ void mlp_layer(const _Float16* src, int sstride, int Kt,
                                          const _Float16* wf, const float* bias,
                                          int Nt, _Float16* dst, int dstride,
                                          bool do_relu, int lane) {
  for (int nt = 0; nt < Nt; ++nt) {
    v8f c = {};
    for (int kt = 0; kt < Kt; ++kt)
      c = wmma_f16(load_A_frag_lds(src, sstride, lane, kt * 32),
                   load_B_frag(wf + (nt * Kt + kt) * 512, lane), c);
    int n = nt * 16 + (lane & 15), g = lane >> 4;
    float bb = bias[n];
#pragma unroll
    for (int r = 0; r < 8; ++r) {
      float v = c[r] + bb;
      if (do_relu) v = v > 0.0f ? v : 0.0f;
      dst[(r + 8 * g) * dstride + n] = (_Float16)v;
    }
  }
}

__global__ void __launch_bounds__(32)
k_head(const float* __restrict__ z_a, const float* __restrict__ z_b,
       const float* __restrict__ h_a, const float* __restrict__ h_b,
       const _Float16* wfF1, const float* bF1, const _Float16* wfF2, const float* bF2,
       const _Float16* wfP1, const float* bP1, const _Float16* wfP2, const float* bP2,
       const _Float16* wfD1, const float* bD1, const _Float16* wfD2, const float* bD2,
       float* __restrict__ preds) {
  __shared__ __align__(16) _Float16 bufIn[16 * 32];
  __shared__ __align__(16) _Float16 bufBig[16 * 136];
  __shared__ __align__(16) _Float16 bufSm[16 * 32];
  int lane = threadIdx.x;
  long tok0 = (long)blockIdx.x * 16;

  for (int idx = lane; idx < 16 * 32; idx += 32) {
    int m = idx >> 5, f = idx & 31;
    long tok = tok0 + m;
    float v = 0.0f;
    if (f < 4)       v = z_a[tok * 4 + f];
    else if (f < 8)  v = z_b[tok * 4 + (f - 4)];
    else if (f < 12) v = h_a[tok * 4 + (f - 8)];
    else if (f < 16) v = h_b[tok * 4 + (f - 12)];
    bufIn[m * 32 + f] = (_Float16)v;
  }
  __syncthreads();
  mlp_layer(bufIn, 32, 1, wfF1, bF1, 2, bufBig, 136, true, lane);  __syncthreads();
  mlp_layer(bufBig, 136, 1, wfF2, bF2, 2, bufSm, 32, false, lane); __syncthreads();
  mlp_layer(bufSm, 32, 1, wfP1, bP1, 8, bufBig, 136, true, lane);  __syncthreads();
  mlp_layer(bufBig, 136, 4, wfP2, bP2, 2, bufIn, 32, false, lane); __syncthreads();
  mlp_layer(bufIn, 32, 1, wfD1, bD1, 8, bufBig, 136, true, lane);  __syncthreads();

  v8f c = {};
#pragma unroll
  for (int kt = 0; kt < 4; ++kt)
    c = wmma_f16(load_A_frag_lds(bufBig, 136, lane, kt * 32),
                 load_B_frag(wfD2 + kt * 512, lane), c);
  int n = lane & 15, g = lane >> 4;
  if (n < 12) {
    float bb = bD2[n];
#pragma unroll
    for (int r = 0; r < 8; ++r) preds[(tok0 + r + 8 * g) * 12 + n] = c[r] + bb;
  }
}

// ---------------------------------------------------------------------------
// Host launcher. Input flattening (depth-first, dict insertion order):
//  0 occ_a 1 occ_b 2 eps_a 3 eps_b | 4-7 enc_a_fast 8-11 enc_b_fast
//  12-15 enc_a_slow 16-19 enc_b_slow | 20-23 gru_a 24-27 gru_b
//  28-31 fusion 32-35 predictor 36-39 decoder | 40 slow_update_every (int)
// d_out: preds(3145728) | kl(1) | mu_a(1048576) | mu_b | h_a | h_b
// ---------------------------------------------------------------------------
extern "C" void kernel_launch(void* const* d_in, const int* in_sizes, int n_in,
                              void* d_out, int out_size, void* d_ws, size_t ws_size,
                              hipStream_t stream) {
  const float* occ_a = (const float*)d_in[0];
  const float* occ_b = (const float*)d_in[1];
  const float* eps_a = (const float*)d_in[2];
  const float* eps_b = (const float*)d_in[3];
  const int*   kptr  = (const int*)d_in[40];

  // Workspace carve-up (~25.3 MB): sized for worst case S == T.
  float* ws_f  = (float*)d_ws;
  float* z_a   = ws_f;
  float* z_b   = z_a   + (size_t)NTOK_ * 4;
  float* sin_a = z_b   + (size_t)NTOK_ * 4;
  float* sin_b = sin_a + (size_t)NTOK_ * 4;
  float* hs_a  = sin_b + (size_t)NTOK_ * 4;
  float* hs_b  = hs_a  + (size_t)NTOK_ * 4;
  _Float16* wf = (_Float16*)(hs_b + (size_t)NTOK_ * 4);

  _Float16* p = wf;
  _Float16 *wf_eaf1 = p; p += 8 * 512;
  _Float16 *wf_eaf2 = p; p += 4 * 512;
  _Float16 *wf_ebf1 = p; p += 8 * 512;
  _Float16 *wf_ebf2 = p; p += 4 * 512;
  _Float16 *wf_eas1 = p; p += 8 * 512;
  _Float16 *wf_eas2 = p; p += 4 * 512;
  _Float16 *wf_ebs1 = p; p += 8 * 512;
  _Float16 *wf_ebs2 = p; p += 4 * 512;
  _Float16 *wf_fu1  = p; p += 2 * 512;
  _Float16 *wf_fu2  = p; p += 2 * 512;
  _Float16 *wf_pr1  = p; p += 8 * 512;
  _Float16 *wf_pr2  = p; p += 8 * 512;
  _Float16 *wf_de1  = p; p += 8 * 512;
  _Float16 *wf_de2  = p; p += 4 * 512;

  float* preds = (float*)d_out;
  float* kl    = preds + (size_t)NTOK_ * 12;
  float* mu_a  = kl + 1;
  float* mu_b  = mu_a + (size_t)NTOK_ * 4;
  float* hA    = mu_b + (size_t)NTOK_ * 4;
  float* hB    = hA   + (size_t)NTOK_ * 4;

  dim3 w32(32);
  // --- weight prep (f32 -> f16 B-fragments, once per call) ---
  k_prep<<<dim3(8, 1), w32, 0, stream>>>((const float*)d_in[4], 128, 12, wf_eaf1, 1);
  k_prep<<<dim3(1, 4), w32, 0, stream>>>((const float*)d_in[6],   8, 128, wf_eaf2, 4);
  k_prep<<<dim3(8, 1), w32, 0, stream>>>((const float*)d_in[8], 128, 12, wf_ebf1, 1);
  k_prep<<<dim3(1, 4), w32, 0, stream>>>((const float*)d_in[10],  8, 128, wf_ebf2, 4);
  k_prep<<<dim3(8, 1), w32, 0, stream>>>((const float*)d_in[12], 128, 12, wf_eas1, 1);
  k_prep<<<dim3(1, 4), w32, 0, stream>>>((const float*)d_in[14],  4, 128, wf_eas2, 4);
  k_prep<<<dim3(8, 1), w32, 0, stream>>>((const float*)d_in[16], 128, 12, wf_ebs1, 1);
  k_prep<<<dim3(1, 4), w32, 0, stream>>>((const float*)d_in[18],  4, 128, wf_ebs2, 4);
  k_prep<<<dim3(2, 1), w32, 0, stream>>>((const float*)d_in[28], 32, 16, wf_fu1, 1);
  k_prep<<<dim3(2, 1), w32, 0, stream>>>((const float*)d_in[30], 32, 32, wf_fu2, 1);
  k_prep<<<dim3(8, 1), w32, 0, stream>>>((const float*)d_in[32], 128, 32, wf_pr1, 1);
  k_prep<<<dim3(2, 4), w32, 0, stream>>>((const float*)d_in[34], 32, 128, wf_pr2, 4);
  k_prep<<<dim3(8, 1), w32, 0, stream>>>((const float*)d_in[36], 128, 32, wf_de1, 1);
  k_prep<<<dim3(1, 4), w32, 0, stream>>>((const float*)d_in[38], 12, 128, wf_de2, 4);

  k_zero<<<1, 1, 0, stream>>>(kl);

  const int tiles = NTOK_ / 16;  // 16384
  // --- fast encoders (+ z, mu, KL) ---
  k_fast_enc<<<tiles, w32, 0, stream>>>(occ_a, eps_a, wf_eaf1, (const float*)d_in[5],
                                        wf_eaf2, (const float*)d_in[7], mu_a, z_a, kl);
  k_fast_enc<<<tiles, w32, 0, stream>>>(occ_b, eps_b, wf_ebf1, (const float*)d_in[9],
                                        wf_ebf2, (const float*)d_in[11], mu_b, z_b, kl);
  // --- slow encoders (grid sized for worst case K=1; extra blocks exit) ---
  k_slow_enc<<<tiles, w32, 0, stream>>>(occ_a, wf_eas1, (const float*)d_in[13],
                                        wf_eas2, (const float*)d_in[15], sin_a, kptr);
  k_slow_enc<<<tiles, w32, 0, stream>>>(occ_b, wf_ebs1, (const float*)d_in[17],
                                        wf_ebs2, (const float*)d_in[19], sin_b, kptr);
  // --- GRU scans ---
  k_gru<<<1, 128, 0, stream>>>(sin_a, (const float*)d_in[20], (const float*)d_in[21],
                               (const float*)d_in[22], (const float*)d_in[23], hs_a, kptr);
  k_gru<<<1, 128, 0, stream>>>(sin_b, (const float*)d_in[24], (const float*)d_in[25],
                               (const float*)d_in[26], (const float*)d_in[27], hs_b, kptr);
  // --- expand slow states into d_out ---
  k_expand<<<(NTOK_ * 4 + 255) / 256, 256, 0, stream>>>(hs_a, hA, kptr);
  k_expand<<<(NTOK_ * 4 + 255) / 256, 256, 0, stream>>>(hs_b, hB, kptr);
  // --- fusion -> predictor -> decoder ---
  k_head<<<tiles, w32, 0, stream>>>(z_a, z_b, hA, hB,
      wf_fu1, (const float*)d_in[29], wf_fu2, (const float*)d_in[31],
      wf_pr1, (const float*)d_in[33], wf_pr2, (const float*)d_in[35],
      wf_de1, (const float*)d_in[37], wf_de2, (const float*)d_in[39], preds);
}
